// Contraction_62345745268781
// MI455X (gfx1250) — compile-verified, matrix operand order checked by reading, not run on hardware
//
#include <hip/hip_runtime.h>

// CDNA5 / gfx1250, wave32. Uses V_WMMA_F32_16X16X32_BF16.
typedef __attribute__((ext_vector_type(16))) __bf16 v16bf;
typedef __attribute__((ext_vector_type(8)))  float  v8f;

namespace {
constexpr int kL   = 16;    // num_ell
constexpr int kC   = 128;   // channels
constexpr int kB   = 2048;  // atoms
constexpr int kE   = 10;    // elements
constexpr int kP3  = 23;
constexpr int kP2  = 4;
constexpr int kKU3 = kP3 * kL;        // 368 "real" K from U3 path
constexpr int kKT  = 12;              // K tiles of 32 -> K = 384 (368 U3 + 4 U2 + 12 pad)
constexpr int kMT  = 16;              // M tiles of 16 -> M = 256 = L*L (tile mt == w)
constexpr int kNG  = kB * kC / 16;    // 16384 groups of 16 (b,c) columns
constexpr int kFragWords = kMT * kKT * 32 * 8;   // 49152 u32 = 192 KB of A fragments
}

union BfPack { unsigned u; __bf16 h[2]; };

// WMMA 16-bit operand lane layout (ISA 7.12.2):
//   lane = {half(1b), idx(4b)}; idx = M row (A) / N col (B); each lane holds 16 bf16.
//   element e of the v16bf maps to K = (e<8 ? 0 : 16) + half*8 + (e&7).
// A fragments are pre-swizzled into LDS in exactly this order so the inner loop
// is a straight 32-byte ds_load per lane per WMMA.

extern "C" __global__ __launch_bounds__(256)
void mace_sym_contraction(const float* __restrict__ x,   // [B,C,L]
                          const float* __restrict__ y,   // [B,E]
                          const float* __restrict__ U3,  // [L,L,L,P3]
                          const float* __restrict__ U2,  // [L,L,P2]
                          const float* __restrict__ U1,  // [L,1]
                          const float* __restrict__ W3,  // [E,P3,C]
                          const float* __restrict__ W2,  // [E,P2,C]
                          const float* __restrict__ W1,  // [E,1,C]
                          float* __restrict__ out) {     // [B,C]
  extern __shared__ unsigned char smem_raw[];
  unsigned*     afw   = (unsigned*)smem_raw;
  const v16bf*  afrag = (const v16bf*)smem_raw;

  const int tid   = threadIdx.x;
  const int lane  = tid & 31;
  const int wid   = tid >> 5;
  const int half  = lane >> 4;
  const int nlane = lane & 15;

  // ---- Stage 1: build bf16 A fragments (U3|U2 flattened 256x384) in LDS ----
  for (int idx = tid; idx < kFragWords; idx += 256) {
    const int word = idx & 7;
    const int ln   = (idx >> 3) & 31;
    const int tile = idx >> 8;            // mt*kKT + kt
    const int mt   = tile / kKT;          // == w
    const int kt   = tile - mt * kKT;
    const int h8   = (ln >> 4) << 3;
    const int m    = ln & 15;             // == x (second L index)
    BfPack p;
#pragma unroll
    for (int s = 0; s < 2; ++s) {
      const int e = word * 2 + s;
      const int K = kt * 32 + ((e < 8) ? 0 : 16) + h8 + (e & 7);
      float v = 0.0f;
      if (K < kKU3) {                     // K = kk*16 + i
        const int kk = K >> 4, i = K & 15;
        v = U3[((mt * kL + m) * kL + i) * kP3 + kk];
      } else if (K < kKU3 + kP2) {
        v = U2[(mt * kL + m) * kP2 + (K - kKU3)];
      }
      p.h[s] = (__bf16)v;
    }
    afw[idx] = p.u;
  }
  __syncthreads();

  const int nwaves = (int)gridDim.x * 8;
#pragma unroll 1
  for (int g = (int)blockIdx.x * 8 + wid; g < kNG; g += nwaves) {
    const int ncol = g * 16 + nlane;      // flat b*C + c for this lane's column
    const int b    = ncol >> 7;
    const int c    = ncol & (kC - 1);

    // Prefetch next group's x row for this lane (global_prefetch_b8).
    if (g + nwaves < kNG)
      __builtin_prefetch(&x[(g + nwaves) * 16 * kL + nlane * kL], 0, 0);

    // ---- per-column data: x vector, y-weighted W's ----
    float xv[kL];
#pragma unroll
    for (int i = 0; i < kL; ++i) xv[i] = x[ncol * kL + i];

    float yv[kE];
#pragma unroll
    for (int e = 0; e < kE; ++e) yv[e] = y[b * kE + e];

    float wy3[kP3];
#pragma unroll
    for (int k = 0; k < kP3; ++k) {
      float s = 0.0f;
#pragma unroll
      for (int e = 0; e < kE; ++e) s += yv[e] * W3[(e * kP3 + k) * kC + c];
      wy3[k] = s;
    }
    float wy2[kP2];
#pragma unroll
    for (int p = 0; p < kP2; ++p) {
      float s = 0.0f;
#pragma unroll
      for (int e = 0; e < kE; ++e) s += yv[e] * W2[(e * kP2 + p) * kC + c];
      wy2[p] = s;
    }
    float wy1 = 0.0f;
#pragma unroll
    for (int e = 0; e < kE; ++e) wy1 += yv[e] * W1[e * kC + c];

    // ---- Stage 2: B-operand fragments in registers: Z[K, n] = xv[i]*wy3[kk] ----
    v16bf zf[kKT];
#pragma unroll
    for (int kt = 0; kt < kKT; ++kt) {
      v16bf z;
#pragma unroll
      for (int e = 0; e < 16; ++e) {
        const int Kb = kt * 32 + ((e < 8) ? 0 : 16) + (e & 7);   // half==0 K
        float v0, v1;
        { const int K = Kb;
          v0 = (K < kKU3) ? xv[K & 15] * wy3[K >> 4]
             : (K < kKU3 + kP2) ? wy2[K - kKU3] : 0.0f; }
        { const int K = Kb + 8;                                   // half==1 K
          v1 = (K < kKU3) ? xv[K & 15] * wy3[K >> 4]
             : (K < kKU3 + kP2) ? wy2[K - kKU3] : 0.0f; }
        z[e] = (__bf16)(half ? v1 : v0);
      }
      zf[kt] = z;
    }

    // Per-lane x slice for the D-tile reduction (acc[r] holds M = r + 8*half).
    float xh[8];
#pragma unroll
    for (int r = 0; r < 8; ++r) xh[r] = half ? xv[r + 8] : xv[r];

    // ---- Stage 3: 16 M-tiles x 12 K-steps of WMMA + fused epilogue ----
    // mt loop deliberately NOT unrolled: keeps zf (96 VGPRs) + one acc tile
    // live, A-fragments streaming from LDS, and no scratch spills.
    float accOut = 0.0f;
#pragma unroll 1
    for (int mt = 0; mt < kMT; ++mt) {
      v8f acc = {};
#pragma unroll
      for (int kt = 0; kt < kKT; ++kt) {
        v16bf a = afrag[(mt * kKT + kt) * 32 + lane];
        acc = __builtin_amdgcn_wmma_f32_16x16x32_bf16(
            /*neg_a=*/false, a, /*neg_b=*/false, zf[kt],
            /*c_mod=*/(short)0, acc, /*reuse_a=*/false, /*reuse_b=*/false);
      }
      // r[w=mt] = sum_x D[x,n]*xv[x]; lane holds D[x=r+8*half, n] in acc[r]
      float partial = 0.0f;
#pragma unroll
      for (int r = 0; r < 8; ++r) partial += acc[r] * xh[r];
      partial += __shfl_xor(partial, 16, 32);          // combine the two halves
      const float xw = x[ncol * kL + mt];              // == xv[mt], re-read to keep
                                                       // xv[] constant-indexed (L0 hit)
      accOut += (partial + U1[mt] * wy1) * xw;         // nu=1 + final x-dot
    }
    if (lane < 16) out[ncol] = accOut;                  // lanes 16-31 are duplicates
  }
}

extern "C" void kernel_launch(void* const* d_in, const int* in_sizes, int n_in,
                              void* d_out, int out_size, void* d_ws, size_t ws_size,
                              hipStream_t stream) {
  const float* x  = (const float*)d_in[0];
  const float* y  = (const float*)d_in[1];
  const float* U3 = (const float*)d_in[2];
  const float* U2 = (const float*)d_in[3];
  const float* U1 = (const float*)d_in[4];
  const float* W3 = (const float*)d_in[5];
  const float* W2 = (const float*)d_in[6];
  const float* W1 = (const float*)d_in[7];
  float* out = (float*)d_out;

  const size_t ldsBytes = (size_t)kFragWords * sizeof(unsigned);   // 192 KB (WGP has 320 KB)
  (void)hipFuncSetAttribute((const void*)mace_sym_contraction,
                            hipFuncAttributeMaxDynamicSharedMemorySize,
                            (int)ldsBytes);
  dim3 grid(512), block(256);   // 4096 waves, 4 column-groups each
  mace_sym_contraction<<<grid, block, ldsBytes, stream>>>(
      x, y, U3, U2, U1, W3, W2, W1, out);
}
